// TwinGraphEncoder_3951369912811
// MI455X (gfx1250) — matrix operand order repaired and checked by reading, compiled, not run on hardware
//
#include <hip/hip_runtime.h>

typedef _Float16 h16;
typedef __attribute__((ext_vector_type(16))) _Float16 v16h;
typedef __attribute__((ext_vector_type(8)))  _Float16 v8h;
typedef __attribute__((ext_vector_type(8)))  float    v8f;

#define TGE_SLOPE 0.01f
#define TGE_EPS   1e-5f
#define TGE_PAD   128

// ----------------------------- device kernels -----------------------------

__global__ void k_fill(float* __restrict__ p, float v, int n) {
  int i = blockIdx.x * 256 + threadIdx.x;
  if (i < n) p[i] = v;
}

__global__ void k_linear(const float* __restrict__ x, const float* __restrict__ w,
                         const float* __restrict__ b, float* __restrict__ out,
                         int N, int Cin, int Cout) {
  int i = blockIdx.x * 256 + threadIdx.x;
  if (i >= N * Cout) return;
  int n = i / Cout, j = i - n * Cout;
  float a = b[j];
  for (int c = 0; c < Cin; ++c) a += x[n * Cin + c] * w[c * Cout + j];
  out[i] = a;
}

// Stage 1 of deterministic BN reduction: fixed 64 partial blocks.
__global__ void k_bnstats(const float* __restrict__ x, const float* __restrict__ x2,
                          int csplit, float* __restrict__ part, int N, int C, int cshift) {
  __shared__ float s0[256];
  __shared__ float s1[256];
  int t = threadIdx.x;
  int c = t & (C - 1);
  int rs = t >> cshift;
  int R = 256 >> cshift;
  float sum = 0.f, sq = 0.f;
  for (int r = blockIdx.x * R + rs; r < N; r += gridDim.x * R) {
    float v = (c < csplit) ? x[r * csplit + c] : x2[r * (C - csplit) + (c - csplit)];
    sum += v; sq += v * v;
  }
  s0[t] = sum; s1[t] = sq;
  __syncthreads();
  if (rs == 0) {
    for (int j = 1; j < R; ++j) { sum += s0[t + (j << cshift)]; sq += s1[t + (j << cshift)]; }
    part[blockIdx.x * 2 * C + c]     = sum;
    part[blockIdx.x * 2 * C + C + c] = sq;
  }
}

__global__ void k_bnfinal(const float* __restrict__ part, float* __restrict__ sums,
                          int nblk, int C) {
  int c = threadIdx.x;
  if (c >= C) return;
  float s = 0.f, q = 0.f;
  for (int b = 0; b < nblk; ++b) { s += part[b * 2 * C + c]; q += part[b * 2 * C + C + c]; }
  sums[c] = s; sums[C + c] = q;
}

// BN (optional) + LeakyReLU (when BN) + fp32->fp16, writing padded [N+128, C]
// with zero pad rows.  Supports a split source (concat of two 2D arrays).
__global__ void k_actprep(const float* __restrict__ x, const float* __restrict__ x2, int csplit,
                          const float* __restrict__ sums, const float* __restrict__ gamma,
                          const float* __restrict__ beta, int useBn, float invN,
                          h16* __restrict__ dst, int N, int C, int cshift) {
  int i = blockIdx.x * 256 + threadIdx.x;
  int P = N + TGE_PAD;
  if (i >= P * C) return;
  int r = i >> cshift;
  int c = i & (C - 1);
  float v = 0.f;
  if (r < N) {
    v = (c < csplit) ? x[r * csplit + c] : x2[r * (C - csplit) + (c - csplit)];
    if (useBn) {
      float m   = sums[c] * invN;
      float var = sums[C + c] * invN - m * m;
      float sc  = gamma[c] * rsqrtf(var + TGE_EPS);
      v = (v - m) * sc + beta[c];
      v = (v >= 0.f) ? v : TGE_SLOPE * v;
    }
  }
  dst[i] = (h16)v;
}

// w[Cout][Cin][9] fp32  ->  wt[Cout][K] fp16 with K index = k*Cin + c
// (B fragments become contiguous 32B per lane).
__global__ void k_wprep(const float* __restrict__ w, h16* __restrict__ wt, int Cout, int Cin) {
  int K = Cin * 9;
  int i = blockIdx.x * 256 + threadIdx.x;
  if (i >= Cout * K) return;
  int o = i / K, kk = i - o * K;
  int knb = kk / Cin, c = kk - knb * Cin;
  wt[i] = (h16)w[(o * Cin + c) * 9 + knb];
}

// Gather-fused WMMA GEMM:
//   out[M,Cout] = gather(act, nbr) x wt^T  (+ bias, + optional residual)
// Block: 256 thr = 8 waves; tile 128(M) x 64(N); wave = 16 rows x 4 n-tiles.
__global__ __launch_bounds__(256) void k_faceconv(
    const h16* __restrict__ act, const h16* __restrict__ wt,
    const int* __restrict__ nbr, const float* __restrict__ bias,
    const float* __restrict__ resid, float* __restrict__ out,
    int Cin, int cinShift, int Cout, int M) {
  const int K    = Cin * 9;
  const int lane = threadIdx.x & 31;
  const int wave = threadIdx.x >> 5;
  const int ln16 = lane & 15;
  const int hi   = lane >> 4;                 // half-wave select
  const int row  = blockIdx.x * 128 + wave * 16 + ln16;
  const int colBase = blockIdx.y * 64;

  v8f acc[4] = {};
  const long r9  = (long)row * 9;
  const int aSel = hi << 4;                   // K offsets 0-15 vs 16-31
  const int bSel = hi << 4;                   // K rows 0-15 vs 16-31
  const h16* wcol = wt + (long)(colBase + ln16) * K + bSel;

  const int nchunk = K >> 5;
  for (int q = 0; q < nchunk; ++q) {
    const int kk = q << 5;
    const int nb = kk >> cinShift;            // neighbor index (Cin is pow2 >= 32)
    const int cb = kk & (Cin - 1);            // channel base inside that neighbor
    const int rp = nbr[r9 + nb];              // gathered padded-row pointer
    const h16* ap = act + ((long)rp << cinShift) + cb + aSel;
    v8h alo = *(const v8h*)ap;                // global_load_b128
    v8h ahi = *(const v8h*)(ap + 8);          // global_load_b128
    v16h afrag = __builtin_shufflevector(alo, ahi,
        0,1,2,3,4,5,6,7,8,9,10,11,12,13,14,15);
#pragma unroll
    for (int t = 0; t < 4; ++t) {
      v16h bfrag = *(const v16h*)(wcol + (long)t * 16 * K + kk);  // 32B contiguous
      acc[t] = __builtin_amdgcn_wmma_f32_16x16x32_f16(
          false, afrag, false, bfrag, (short)0, acc[t], false, false);
    }
  }

  const int orow0 = blockIdx.x * 128 + wave * 16 + (hi << 3);
#pragma unroll
  for (int t = 0; t < 4; ++t) {
    const int col = colBase + (t << 4) + ln16;
    const float bv = bias[col];
#pragma unroll
    for (int v = 0; v < 8; ++v) {
      const long oi = (long)(orow0 + v) * Cout + col;
      float val = acc[t][v] + bv;
      if (resid) val += resid[oi];
      out[oi] = val;
    }
  }
}

// ------------------------------- pooling ----------------------------------

__global__ void k_minred(const float* __restrict__ x, int n, float* __restrict__ out) {
  __shared__ float s[256];
  float m = 3.0e38f;
  for (int i = blockIdx.x * 256 + threadIdx.x; i < n; i += gridDim.x * 256)
    m = fminf(m, x[i]);
  s[threadIdx.x] = m;
  __syncthreads();
  for (int o = 128; o > 0; o >>= 1) {
    if (threadIdx.x < o) s[threadIdx.x] = fminf(s[threadIdx.x], s[threadIdx.x + o]);
    __syncthreads();
  }
  if (threadIdx.x == 0) {
    float v = s[0];
    int* a = (int*)out;
    int old = *a;
    while (__int_as_float(old) > v) {
      int assumed = old;
      old = atomicCAS(a, assumed, __float_as_int(v));
      if (old == assumed) break;
    }
  }
}

__global__ void k_poolfill(float* __restrict__ p, const float* __restrict__ minv, int n) {
  int i = blockIdx.x * 256 + threadIdx.x;
  if (i < n) p[i] = minv[0] - 0.001f;
}

__global__ void k_poolscatter(const float* __restrict__ x, const int* __restrict__ pmap,
                              float* __restrict__ p, int N, int C) {
  int i = blockIdx.x * 256 + threadIdx.x;
  if (i >= N * C) return;
  int r = i / C, c = i - r * C;
  float v = x[i];
  int* a = (int*)&p[(long)pmap[r] * C + c];
  int old = *a;
  while (__int_as_float(old) < v) {             // exact max: order-independent
    int assumed = old;
    old = atomicCAS(a, assumed, __float_as_int(v));
    if (old == assumed) break;
  }
}

__global__ void k_latgather(const float* __restrict__ p, const int* __restrict__ lat,
                            float* __restrict__ tmp, int L, int C) {
  int i = blockIdx.x * 256 + threadIdx.x;
  if (i >= L * C) return;
  int j = i / C, c = i - j * C;
  tmp[i] = p[(long)lat[j * 2 + 1] * C + c];
}

__global__ void k_latscatter(float* __restrict__ p, const int* __restrict__ lat,
                             const float* __restrict__ tmp, int L, int C) {
  int i = blockIdx.x * 256 + threadIdx.x;
  if (i >= L * C) return;
  int j = i / C, c = i - j * C;
  p[(long)lat[j * 2 + 0] * C + c] = tmp[i];
}

// ------------------------------ host side ---------------------------------

struct ResP {
  const float *b0, *b1, *be0, *be1, *bs, *g0, *g1, *w0, *w1, *ws;
  int cin, cout; bool has_s;
};

// jax pytree leaf order for a res dict (keys sorted):
// b0,b1,be0,be1,[bs],g0,g1,w0,w1,[ws]
static ResP get_res(void* const* d, int& idx, int cin, int cout) {
  ResP r; r.cin = cin; r.cout = cout; r.has_s = (cin != cout);
  r.b0 = (const float*)d[idx++]; r.b1 = (const float*)d[idx++];
  r.be0 = (const float*)d[idx++]; r.be1 = (const float*)d[idx++];
  r.bs = nullptr; r.ws = nullptr;
  if (r.has_s) r.bs = (const float*)d[idx++];
  r.g0 = (const float*)d[idx++]; r.g1 = (const float*)d[idx++];
  r.w0 = (const float*)d[idx++]; r.w1 = (const float*)d[idx++];
  if (r.has_s) r.ws = (const float*)d[idx++];
  return r;
}

struct Ctx {
  hipStream_t s;
  float* SUMS; float* MINV; float* PART; float* LATT;
  h16* WT; h16* ACT;
};

static inline int cdiv(long a, int b) { return (int)((a + (long)b - 1) / b); }
static inline int ilog2i(int x) { int s = 0; while ((1 << s) < x) ++s; return s; }

static void bn_act(Ctx& c, const float* X, const float* X2, int csplit,
                   const float* g, const float* be, bool bn, int N, int C) {
  int cs = ilog2i(C);
  if (bn) {
    k_bnstats<<<64, 256, 0, c.s>>>(X, X2, csplit, c.PART, N, C, cs);
    k_bnfinal<<<1, 256, 0, c.s>>>(c.PART, c.SUMS, 64, C);
  }
  long total = (long)(N + TGE_PAD) * C;
  k_actprep<<<cdiv(total, 256), 256, 0, c.s>>>(X, X2, csplit, c.SUMS, g, be,
                                               bn ? 1 : 0, 1.0f / (float)N,
                                               c.ACT, N, C, cs);
}

static void face_conv(Ctx& c, const float* w, int cin, int cout, const int* nbr,
                      const float* bias, const float* resid, float* out, int M) {
  int K = cin * 9;
  k_wprep<<<cdiv((long)cout * K, 256), 256, 0, c.s>>>(w, c.WT, cout, cin);
  dim3 grid(M / 128, cout / 64);
  k_faceconv<<<grid, 256, 0, c.s>>>(c.ACT, c.WT, nbr, bias, resid, out,
                                    cin, ilog2i(cin), cout, M);
}

static void res_block(Ctx& c, const ResP& p, const float* X, const float* X2, int csplit,
                      int N, const int* nbr, float* H1, float* OUT) {
  bn_act(c, p.has_s ? X : X, X2, csplit, p.g0, p.be0, true, N, p.cin);
  face_conv(c, p.w0, p.cin, p.cout, nbr, p.b0, nullptr, H1, N);
  bn_act(c, H1, nullptr, p.cout, p.g1, p.be1, true, N, p.cout);
  face_conv(c, p.w1, p.cout, p.cout, nbr, p.b1, p.has_s ? nullptr : X, OUT, N);
  if (p.has_s) {                                  // nin_shortcut: OUT += conv(X)+bs
    bn_act(c, X, X2, csplit, nullptr, nullptr, false, N, p.cin);
    face_conv(c, p.ws, p.cin, p.cout, nbr, p.bs, OUT, OUT, N);
  }
}

static void pool_max(Ctx& c, const float* x, int N, int C, const int* pmap,
                     const int* lat, int L, float* out, int Nout) {
  int n = N * C;
  k_fill<<<1, 1, 0, c.s>>>(c.MINV, 3.0e38f, 1);
  int mb = cdiv((long)n, 256); if (mb > 512) mb = 512;
  k_minred<<<mb, 256, 0, c.s>>>(x, n, c.MINV);
  k_poolfill<<<cdiv((long)Nout * C, 256), 256, 0, c.s>>>(out, c.MINV, Nout * C);
  k_poolscatter<<<cdiv((long)n, 256), 256, 0, c.s>>>(x, pmap, out, N, C);
  k_latgather<<<cdiv((long)L * C, 256), 256, 0, c.s>>>(out, lat, c.LATT, L, C);
  k_latscatter<<<cdiv((long)L * C, 256), 256, 0, c.s>>>(out, lat, c.LATT, L, C);
}

extern "C" void kernel_launch(void* const* d_in, const int* in_sizes, int n_in,
                              void* d_out, int out_size, void* d_ws, size_t ws_size,
                              hipStream_t stream) {
  (void)out_size; (void)ws_size;
  const int N0 = 131072, N1 = 32768, N2 = 8192, N3 = 2048, N4 = 512, N5 = 128;

  const float* x0in = (const float*)d_in[0];
  const float* x1in = (const float*)d_in[1];

  // params pytree (top-level sorted): d00,d0b1,d10,down[0..3],lin0_b,lin0_w,
  // lin1_b,lin1_w,mid,out_b,out_be,out_g,out_w
  int idx = 2;
  ResP d00  = get_res(d_in, idx, 32, 64);
  ResP d0b1 = get_res(d_in, idx, 128, 64);
  ResP d10  = get_res(d_in, idx, 32, 64);
  ResP dn0  = get_res(d_in, idx, 64, 128);
  ResP dn1  = get_res(d_in, idx, 128, 128);
  ResP dn2  = get_res(d_in, idx, 128, 128);
  ResP dn3  = get_res(d_in, idx, 128, 256);
  const float* lin0_b = (const float*)d_in[idx++];
  const float* lin0_w = (const float*)d_in[idx++];
  const float* lin1_b = (const float*)d_in[idx++];
  const float* lin1_w = (const float*)d_in[idx++];
  ResP mid = get_res(d_in, idx, 256, 256);
  const float* out_b  = (const float*)d_in[idx++];
  const float* out_be = (const float*)d_in[idx++];
  const float* out_g  = (const float*)d_in[idx++];
  const float* out_w  = (const float*)d_in[idx++];

  // anchor face_neighborhood by its unique element count (131072*9)
  int fnbIdx = idx;
  for (int i = 2; i < n_in; ++i) if (in_sizes[i] == N0 * 9) { fnbIdx = i; break; }
  const int* fnb = (const int*)d_in[fnbIdx];
  const int* sub[5];
  for (int i = 0; i < 5; ++i) sub[i] = (const int*)d_in[fnbIdx + 1 + i];
  // anchor pool_map0 (count 131072; is_pad* are 131200/32896/... so no clash)
  int pmIdx = fnbIdx + 12;
  for (int i = fnbIdx + 6; i < n_in; ++i) if (in_sizes[i] == N0) { pmIdx = i; break; }
  const int* pmap[5]; const int* lat[5];
  for (int i = 0; i < 5; ++i) pmap[i] = (const int*)d_in[pmIdx + i];
  for (int i = 0; i < 5; ++i) lat[i]  = (const int*)d_in[pmIdx + 5 + i];
  const int Lsz[5] = {2048, 512, 128, 32, 8};

  // ---- workspace carve (bump allocator, 256B aligned) ----
  char* base = (char*)d_ws;
  size_t off = 0;
  auto carve = [&](size_t bytes) -> char* {
    char* p = base + off;
    off += (bytes + 255) & ~(size_t)255;
    return p;
  };
  Ctx c; c.s = stream;
  c.SUMS = (float*)carve(512 * sizeof(float));                   // sum | sumsq
  c.MINV = (float*)carve(256);
  c.PART = (float*)carve(64 * 512 * sizeof(float));              // BN partials
  c.WT   = (h16*)carve((size_t)256 * 2304 * sizeof(h16));        // packed weights
  c.ACT  = (h16*)carve((size_t)(N0 + TGE_PAD) * 128 * sizeof(h16)); // padded f16 acts
  float* LIN0 = (float*)carve((size_t)N0 * 64 * sizeof(float));  // also pooled-x buf
  float* LIN1 = (float*)carve((size_t)N0 * 64 * sizeof(float));  // also mid output
  float* X0B  = (float*)carve((size_t)N0 * 64 * sizeof(float));
  float* X1B  = (float*)carve((size_t)N0 * 64 * sizeof(float));
  float* BUFH = (float*)carve((size_t)N0 * 64 * sizeof(float));  // res-block h1
  c.LATT = (float*)carve(1 << 20);

  // ---- output (tuple concatenated flat) ----
  float* feat0 = (float*)d_out;
  float* feat1 = feat0 + (size_t)N0 * 64;
  float* feat2 = feat1 + (size_t)N1 * 128;
  float* feat3 = feat2 + (size_t)N2 * 128;
  float* feat4 = feat3 + (size_t)N3 * 128;
  float* feat5 = feat4 + (size_t)N4 * 256;

  // ---- stem ----
  k_linear<<<cdiv((long)N0 * 32, 256), 256, 0, stream>>>(x0in, lin0_w, lin0_b, LIN0, N0, 3, 32);
  k_linear<<<cdiv((long)N0 * 32, 256), 256, 0, stream>>>(x1in, lin1_w, lin1_b, LIN1, N0, 7, 32);
  res_block(c, d00, LIN0, nullptr, 32, N0, fnb, BUFH, X0B);
  res_block(c, d10, LIN1, nullptr, 32, N0, fnb, BUFH, X1B);
  res_block(c, d0b1, X0B, X1B, 64, N0, fnb, BUFH, feat0);        // concat handled via split

  // ---- down path ----
  float* xa = LIN0;
  pool_max(c, feat0, N0, 64,  pmap[0], lat[0], Lsz[0], xa, N1);
  res_block(c, dn0, xa, nullptr, 64,  N1, sub[0], BUFH, feat1);
  pool_max(c, feat1, N1, 128, pmap[1], lat[1], Lsz[1], xa, N2);
  res_block(c, dn1, xa, nullptr, 128, N2, sub[1], BUFH, feat2);
  pool_max(c, feat2, N2, 128, pmap[2], lat[2], Lsz[2], xa, N3);
  res_block(c, dn2, xa, nullptr, 128, N3, sub[2], BUFH, feat3);
  pool_max(c, feat3, N3, 128, pmap[3], lat[3], Lsz[3], xa, N4);
  res_block(c, dn3, xa, nullptr, 128, N4, sub[3], BUFH, feat4);

  // ---- mid + output head ----
  pool_max(c, feat4, N4, 256, pmap[4], lat[4], Lsz[4], xa, N5);
  res_block(c, mid, xa, nullptr, 256, N5, sub[4], BUFH, LIN1);
  bn_act(c, LIN1, nullptr, 256, out_g, out_be, true, N5, 256);
  face_conv(c, out_w, 256, 256, sub[4], out_b, nullptr, feat5, N5);
}
